// ConvPE_73675868996363
// MI455X (gfx1250) — matrix operand order repaired and checked by reading, not compiled, not run on hardware
//
#include <hip/hip_runtime.h>

// ---------------------------------------------------------------------------
// Fused Conv1d(k=3) + BatchNorm(inference) + 4-step LIF + residual for MI455X.
// bf16 WMMA (v_wmma_f32_16x16x32_bf16) with fp32 accumulation.
// Staging path vectorized: global_load_b128 (float4) -> packed bf16 ds_store_b64.
// ---------------------------------------------------------------------------

typedef __attribute__((ext_vector_type(16))) __bf16          v16bf;
typedef __attribute__((ext_vector_type(8)))  float           v8f;
typedef __attribute__((ext_vector_type(4)))  float           f4;
typedef __attribute__((ext_vector_type(8)))  unsigned short  ush8;
typedef __attribute__((ext_vector_type(4)))  unsigned short  ush4;

#define L_DIM   2048
#define D_DIM   128
#define T_STEPS 4
#define B_DIM   32          // TB / T
#define TL      32          // l-positions per workgroup
#define XROWS   (TL + 2)    // +2 halo rows (conv pad=1)
#define XPITCH  (D_DIM + 8) // pad LDS rows -> 272B stride, conflict-free 16B reads

__device__ __forceinline__ unsigned short f2bf(float f) {
    unsigned int u = __float_as_uint(f);
    u += 0x7FFFu + ((u >> 16) & 1u);          // round-to-nearest-even
    return (unsigned short)(u >> 16);
}

union Frag16 { ush8 h[2]; v16bf v; };

// Pack conv_w (O=128, I=128, K=3) fp32 -> bf16 in WMMA-B lane order:
// wpk[((tap*4 + chunk)*128 + o)*32 + j] = bf16(w[o][chunk*32 + j][tap])
// so a lane's 16 B-fragment halves are one contiguous 32B block.
__global__ void pack_weights(const float* __restrict__ w,
                             unsigned short* __restrict__ wpk) {
    int idx   = blockIdx.x * 256 + threadIdx.x;   // 3*4*128*32 = 49152 total
    int j     = idx & 31;
    int o     = (idx >> 5) & 127;
    int chunk = (idx >> 12) & 3;
    int tap   = idx >> 14;
    wpk[idx] = f2bf(w[o * 384 + (chunk * 32 + j) * 3 + tap]);
}

__global__ __launch_bounds__(256)
void conv_bn_lif_kernel(const float* __restrict__ x,
                        const unsigned short* __restrict__ wpk,
                        const float* __restrict__ gamma,
                        const float* __restrict__ beta,
                        const float* __restrict__ mean,
                        const float* __restrict__ var,
                        float* __restrict__ out) {
    __shared__ unsigned short sX[T_STEPS][XROWS][XPITCH];
    __shared__ float sScale[D_DIM];
    __shared__ float sShift[D_DIM];

    const int tid  = threadIdx.x;
    const int wave = tid >> 5;
    const int lane = tid & 31;
    const int l0   = blockIdx.x * TL;
    const int b    = blockIdx.y;

    if (tid < D_DIM) {
        float sc = gamma[tid] * rsqrtf(var[tid] + 1e-5f);
        sScale[tid] = sc;
        sShift[tid] = beta[tid] - mean[tid] * sc;
    }

    // Stage x (L, TB, D) fp32 -> bf16 LDS tiles for the 4 time steps of batch b.
    // float4 loads: channels contiguous per l-row -> 512B per wave.
    for (int idx = tid; idx < T_STEPS * XROWS * (D_DIM / 4); idx += 256) {
        int t    = idx / (XROWS * 32);
        int rem  = idx - t * (XROWS * 32);
        int lrow = rem >> 5;
        int i4   = (rem & 31) << 2;
        int gl   = l0 - 1 + lrow;
        f4 v = {};
        if (gl >= 0 && gl < L_DIM)
            v = *(const f4*)(x + (gl << 14) + ((t * B_DIM + b) << 7) + i4);
        ush4 h;
        h.x = f2bf(v.x); h.y = f2bf(v.y); h.z = f2bf(v.z); h.w = f2bf(v.w);
        *(ush4*)&sX[t][lrow][i4] = h;
    }
    __syncthreads();

    const int hi  = (lane >= 16) ? 1 : 0;
    const int lid = lane & 15;
    const v8f zero = {};

    #pragma unroll
    for (int cc = 0; cc < 2; ++cc) {
        const int m_sub = cc;
        const int n_sub = wave;
        const int o     = n_sub * 16 + lid;      // output channel of this lane

        v8f acc[T_STEPS];
        #pragma unroll
        for (int t = 0; t < T_STEPS; ++t) acc[t] = zero;

        #pragma unroll
        for (int tap = 0; tap < 3; ++tap) {
            #pragma unroll
            for (int chunk = 0; chunk < 4; ++chunk) {
                // B fragment: K(e) = hi*16 + e, N = lid  (32B contiguous)
                Frag16 bf;
                const ush8* bp = (const ush8*)(wpk +
                    ((((tap * 4 + chunk) * 128) + o) * 32 + hi * 16));
                bf.h[0] = bp[0];
                bf.h[1] = bp[1];

                // A fragment: M = lid, K(e) = hi*16 + e
                const int i0 = chunk * 32 + hi * 16;
                #pragma unroll
                for (int t = 0; t < T_STEPS; ++t) {
                    const int lrow = m_sub * 16 + lid + tap; // includes -1 halo
                    Frag16 af;
                    af.h[0] = *(const ush8*)&sX[t][lrow][i0];
                    af.h[1] = *(const ush8*)&sX[t][lrow][i0 + 8];
                    acc[t] = __builtin_amdgcn_wmma_f32_16x16x32_bf16(
                        false, af.v, false, bf.v, (short)0, acc[t],
                        false, false);
                }
            }
        }

        // BN + LIF recurrence over t + residual store.
        const float scale = sScale[o];
        const float shift = sShift[o];
        #pragma unroll
        for (int r = 0; r < 8; ++r) {
            const int l = l0 + m_sub * 16 + hi * 8 + r;  // C/D: M = r + 8*hi
            float v = 0.0f;
            #pragma unroll
            for (int t = 0; t < T_STEPS; ++t) {
                float y = acc[t][r] * scale + shift;     // conv + BN
                float h = 0.5f * (v + y);                // charge, tau = 2
                float s = (h >= 1.0f) ? 1.0f : 0.0f;     // Heaviside spike
                v = h * (1.0f - s);                      // hard reset to 0
                int oidx = (l << 14) + ((t * B_DIM + b) << 7) + o;
                out[oidx] = x[oidx] + s;                 // residual
            }
        }
    }
}

extern "C" void kernel_launch(void* const* d_in, const int* in_sizes, int n_in,
                              void* d_out, int out_size, void* d_ws, size_t ws_size,
                              hipStream_t stream) {
    const float* x     = (const float*)d_in[0];
    const float* w     = (const float*)d_in[1];
    const float* gamma = (const float*)d_in[2];
    const float* beta  = (const float*)d_in[3];
    const float* mean  = (const float*)d_in[4];
    const float* var   = (const float*)d_in[5];
    float* out = (float*)d_out;
    unsigned short* wpk = (unsigned short*)d_ws;   // 96 KB bf16 packed weights

    pack_weights<<<192, 256, 0, stream>>>(w, wpk);

    dim3 grid(L_DIM / TL, B_DIM);                  // 64 x 32 = 2048 workgroups
    conv_bn_lif_kernel<<<grid, 256, 0, stream>>>(x, wpk, gamma, beta,
                                                 mean, var, out);
}
